// TransformerDecoderLayer_90881507984175
// MI455X (gfx1250) — compile-verified, matrix operand order checked by reading, not run on hardware
//
#include <hip/hip_runtime.h>

// ---------------------------------------------------------------------------
// CDNA5 WMMA types / helpers
// ---------------------------------------------------------------------------
typedef __bf16 v16bf __attribute__((ext_vector_type(16)));
typedef float  v8f   __attribute__((ext_vector_type(8)));
typedef int    v4i   __attribute__((ext_vector_type(4)));

__device__ __forceinline__ v8f wmma_bf16(v16bf a, v16bf b, v8f c) {
  return __builtin_amdgcn_wmma_f32_16x16x32_bf16(false, a, false, b, (short)0, c,
                                                 false, false);
}

__device__ __forceinline__ unsigned short f32_to_bf16(float f) {
  union { float f; unsigned u; } x; x.f = f;
  unsigned u = x.u;
  unsigned r = 0x7FFFu + ((u >> 16) & 1u);   // round-to-nearest-even
  return (unsigned short)((u + r) >> 16);
}

// Swizzled (fragment-major) LDS layout for one 16x32 (MxK) / 32x16 (KxN) tile:
// element with contraction index k (0..31) for lane-column c (0..15) sits at
//   ((k>>3 & 1)*16 + c)*16 + ((k&7) | ((k&16)>>1))
// so lane L of a wave reads its 16 fragment values as 32 contiguous bytes at
// tile_base + L*16 (two ds_load_b128).
__device__ __forceinline__ int swz(int k, int c) {
  int half = (k >> 3) & 1;
  int j = (k & 7) | ((k & 16) >> 1);
  return (((half << 4) + c) << 4) + j;
}

__device__ __forceinline__ v16bf ld_frag(const unsigned short* p) {
  return *reinterpret_cast<const v16bf*>(p);  // 32B -> 2x ds_load_b128
}

// ---------------------------------------------------------------------------
// Async global->LDS copy of 16 bytes (CDNA5 ASYNCcnt path), with fallback.
// Probed signature: arg0 is `int4 __device__*` (AS1); pair with AS3 for LDS.
// ---------------------------------------------------------------------------
#if defined(__has_builtin)
#if __has_builtin(__builtin_amdgcn_global_load_async_to_lds_b128)
#define HAVE_ASYNC_LDS 1
#endif
#endif
#ifndef HAVE_ASYNC_LDS
#define HAVE_ASYNC_LDS 0
#endif

__device__ __forceinline__ void copy16_g2l(const unsigned short* g,
                                           unsigned short* l) {
#if HAVE_ASYNC_LDS
  typedef __attribute__((address_space(1))) v4i GV4;
  typedef __attribute__((address_space(3))) v4i LV4;
  __builtin_amdgcn_global_load_async_to_lds_b128((GV4*)(unsigned short*)g,
                                                 (LV4*)l, 0, 0);
#else
  *reinterpret_cast<uint4*>(l) = *reinterpret_cast<const uint4*>(g);
#endif
}

__device__ __forceinline__ void async_fence() {
#if HAVE_ASYNC_LDS
  asm volatile("s_wait_asynccnt 0" ::: "memory");
#endif
}

// ---------------------------------------------------------------------------
// fp32 -> bf16 convert (grid-stride)
// ---------------------------------------------------------------------------
__global__ void cvt_bf16_kernel(const float* __restrict__ in,
                                unsigned short* __restrict__ out, size_t n) {
  size_t i = (size_t)blockIdx.x * blockDim.x + threadIdx.x;
  size_t stride = (size_t)gridDim.x * blockDim.x;
  for (; i < n; i += stride) out[i] = f32_to_bf16(in[i]);
}

// ---------------------------------------------------------------------------
// GEMM: C(MxN,f32) = A(MxK,bf16,row) x B(KxN,bf16,row) + bias[N], opt ReLU,
// optional bf16 copy. Block tile 128x128, 8 waves, wave tile 32x64
// (8 wmma/K-step/wave), K-step 32, swizzled LDS staging.
// ---------------------------------------------------------------------------
__global__ __launch_bounds__(256) void gemm_bf16_kernel(
    const unsigned short* __restrict__ A, const unsigned short* __restrict__ B,
    const float* __restrict__ bias, float* __restrict__ C32,
    unsigned short* __restrict__ C16, int M, int N, int K, int relu) {
  __shared__ __align__(32) unsigned short sA[8 * 512];  // 128x32, frag-major
  __shared__ __align__(32) unsigned short sB[8 * 512];  // 32x128, frag-major

  const int tid  = threadIdx.x;
  const int lane = tid & 31;
  const int wid  = tid >> 5;
  const int half = (lane >> 4) & 1;
  const int lm   = lane & 15;
  const int bm = blockIdx.y * 128;
  const int bn = blockIdx.x * 128;
  const int wm = (wid >> 1) * 32;
  const int wn = (wid & 1) * 64;

  v8f acc[2][4] = {};

  for (int kk = 0; kk < K; kk += 32) {
    // A tile 128x32: contraction (K) is the global fast axis -> 16B-contiguous
    // in the swizzled layout; async copy when available.
    for (int q = tid; q < 512; q += 256) {
      int r = q >> 2, c0 = (q & 3) * 8;
      copy16_g2l(A + (size_t)(bm + r) * K + kk + c0,
                 sA + ((r >> 4) << 9) + swz(c0, r & 15));
    }
    // B tile 32x128: global fast axis is N -> vector load + 8-way LDS scatter.
    for (int q = tid; q < 512; q += 256) {
      int r = q >> 4, c0 = (q & 15) * 8;
      uint4 d = *reinterpret_cast<const uint4*>(B + (size_t)(kk + r) * N + bn + c0);
      const unsigned short* e = (const unsigned short*)&d;
      int base = (c0 >> 4) << 9;
      for (int t = 0; t < 8; ++t) sB[base + swz(r, (c0 & 15) + t)] = e[t];
    }
    async_fence();
    __syncthreads();

    if (kk + 32 < K)  // prefetch next A tile -> global_prefetch_b8
      __builtin_prefetch(A + (size_t)(bm + (tid >> 1)) * K + kk + 32, 0, 1);

    v16bf af[2], bfr[4];
    af[0] = ld_frag(sA + (((wm >> 4) + 0) << 9) + lane * 16);
    af[1] = ld_frag(sA + (((wm >> 4) + 1) << 9) + lane * 16);
    for (int b = 0; b < 4; ++b)
      bfr[b] = ld_frag(sB + (((wn >> 4) + b) << 9) + lane * 16);

    for (int a = 0; a < 2; ++a)
      for (int b = 0; b < 4; ++b)
        acc[a][b] = wmma_bf16(af[a], bfr[b], acc[a][b]);
    __syncthreads();
  }

  for (int a = 0; a < 2; ++a)
    for (int b = 0; b < 4; ++b)
      for (int i = 0; i < 8; ++i) {
        int row = bm + wm + a * 16 + i + 8 * half;  // C/D layout (ISA 7.12.2)
        int col = bn + wn + b * 16 + lm;
        float v = acc[a][b][i] + (bias ? bias[col] : 0.f);
        if (relu) v = fmaxf(v, 0.f);
        size_t off = (size_t)row * N + col;
        if (C32) C32[off] = v;
        if (C16) C16[off] = f32_to_bf16(v);
      }
}

// ---------------------------------------------------------------------------
// Flash attention (bf16 in/out, fp32 accum). One workgroup = (b, h, 64 q rows),
// 4 waves x 16 q rows, K/V streamed in 32-key chunks through swizzled LDS.
// Online softmax with half-wave shuffle row reductions. bias: (H, Lq, Lk) f32.
// Q/K/V layout: (B*Lrows, 1024), head h at columns h*64..h*64+63.
// ---------------------------------------------------------------------------
template <bool CAUSAL>
__global__ __launch_bounds__(128) void flash_attn_kernel(
    const unsigned short* __restrict__ Qb, const unsigned short* __restrict__ Kb,
    const unsigned short* __restrict__ Vb, const float* __restrict__ bias,
    unsigned short* __restrict__ Ob, int Lq, int Lk, float inv_temp) {
  __shared__ __align__(32) unsigned short sK[4 * 512];     // [c][ntile] frag-major
  __shared__ __align__(32) unsigned short sV[4 * 512];     // [dtile] frag-major
  __shared__ __align__(32) unsigned short sQP[4][2 * 512]; // per-wave Q, then P

  const int tid  = threadIdx.x;
  const int lane = tid & 31;
  const int w    = tid >> 5;
  const int half = (lane >> 4) & 1;
  const int lm   = lane & 15;
  const int qblk = blockIdx.x;
  const int h    = blockIdx.y;
  const int bz   = blockIdx.z;
  const int qbase = qblk * 64 + w * 16;
  const int D = 1024, HD = 64;

  // Stage this wave's Q tile (16x64) frag-major; contraction (head dim) is the
  // global fast axis -> contiguous 16B copies.
  {
    const unsigned short* Qrow = Qb + ((size_t)(bz * Lq + qbase)) * D + h * HD;
    for (int q = lane; q < 128; q += 32) {
      int r = q >> 3, d0 = (q & 7) * 8;
      copy16_g2l(Qrow + (size_t)r * D + d0,
                 sQP[w] + ((d0 >> 5) << 9) + swz(d0 & 31, r));
    }
  }
  async_fence();
  __syncthreads();
  v16bf qf[2];
  qf[0] = ld_frag(sQP[w] + lane * 16);
  qf[1] = ld_frag(sQP[w] + 512 + lane * 16);
  __syncthreads();

  v8f o0 = {}, o1 = {}, o2 = {}, o3 = {};
  float m[8], l[8];
  for (int i = 0; i < 8; ++i) { m[i] = -1e30f; l[i] = 0.f; }

  const int nchunks = CAUSAL ? (2 * qblk + 2) : (Lk >> 5);
  for (int kc = 0; kc < nchunks; ++kc) {
    const int k0 = kc << 5;
    // Stage 32 keys of K and V. K: contraction = head dim (fast axis) ->
    // contiguous async copies. V: contraction = key -> vector load + scatter.
    for (int q = tid; q < 256; q += 128) {
      int key = q >> 3, d0 = (q & 7) * 8;
      size_t goff = ((size_t)(bz * Lk + k0 + key)) * D + h * HD + d0;
      copy16_g2l(Kb + goff,
                 sK + ((d0 >> 5) << 10) + ((key >> 4) << 9) + swz(d0 & 31, key & 15));
      uint4 vv = *reinterpret_cast<const uint4*>(Vb + goff);
      const unsigned short* e = (const unsigned short*)&vv;
      int vbase = (d0 >> 4) << 9;
      for (int t = 0; t < 8; ++t) sV[vbase + swz(key, (d0 & 15) + t)] = e[t];
    }
    async_fence();
    __syncthreads();

    // S = Q K^T for 32 keys -> two 16x16 f32 accumulators.
    v8f s0 = {}, s1 = {};
    for (int c = 0; c < 2; ++c) {
      v16bf kf0 = ld_frag(sK + (c << 10) + lane * 16);
      v16bf kf1 = ld_frag(sK + (c << 10) + 512 + lane * 16);
      s0 = wmma_bf16(qf[c], kf0, s0);
      s1 = wmma_bf16(qf[c], kf1, s1);
    }

    // Online softmax; emit P (bf16) directly in fragment-major order.
    for (int i = 0; i < 8; ++i) {
      int row = i + 8 * half;
      int q = qbase + row;
      int key0 = k0 + lm, key1 = k0 + 16 + lm;
      size_t brow = ((size_t)h * Lq + q) * Lk;
      float a0 = s0[i] * inv_temp + bias[brow + key0];
      float a1 = s1[i] * inv_temp + bias[brow + key1];
      if (CAUSAL) {
        if (key0 > q) a0 = -1e30f;
        if (key1 > q) a1 = -1e30f;
      }
      float cm = fmaxf(a0, a1);
      for (int off = 8; off > 0; off >>= 1)
        cm = fmaxf(cm, __shfl_xor(cm, off, 16));
      float nm = fmaxf(m[i], cm);
      float alpha = __expf(m[i] - nm);
      float p0 = __expf(a0 - nm);
      float p1 = __expf(a1 - nm);
      m[i] = nm;
      l[i] = l[i] * alpha + p0 + p1;
      o0[i] *= alpha; o1[i] *= alpha; o2[i] *= alpha; o3[i] *= alpha;
      sQP[w][swz(lm, row)]      = f32_to_bf16(p0);
      sQP[w][swz(16 + lm, row)] = f32_to_bf16(p1);
    }
    asm volatile("s_wait_dscnt 0" ::: "memory");  // intra-wave LDS round trip

    v16bf pf = ld_frag(sQP[w] + lane * 16);
    o0 = wmma_bf16(pf, ld_frag(sV + lane * 16), o0);
    o1 = wmma_bf16(pf, ld_frag(sV + 512 + lane * 16), o1);
    o2 = wmma_bf16(pf, ld_frag(sV + 1024 + lane * 16), o2);
    o3 = wmma_bf16(pf, ld_frag(sV + 1536 + lane * 16), o3);
    __syncthreads();
  }

  for (int i = 0; i < 8; ++i) {
    float li = l[i];
    for (int off = 8; off > 0; off >>= 1) li += __shfl_xor(li, off, 16);
    float inv = 1.f / li;
    int q = qbase + i + 8 * half;
    size_t ob = ((size_t)(bz * Lq + q)) * D + h * HD;
    Ob[ob + lm]      = f32_to_bf16(o0[i] * inv);
    Ob[ob + 16 + lm] = f32_to_bf16(o1[i] * inv);
    Ob[ob + 32 + lm] = f32_to_bf16(o2[i] * inv);
    Ob[ob + 48 + lm] = f32_to_bf16(o3[i] * inv);
  }
}

// ---------------------------------------------------------------------------
// y = LayerNorm(x + res) * g + b over last dim (D=1024). 1 block / row.
// ---------------------------------------------------------------------------
__global__ __launch_bounds__(256) void add_ln_kernel(
    const float* __restrict__ x, const float* __restrict__ res,
    const float* __restrict__ g, const float* __restrict__ b,
    float* __restrict__ y, unsigned short* __restrict__ ybf) {
  __shared__ float red[256];
  const int tid = threadIdx.x;
  const size_t base = (size_t)blockIdx.x * 1024;

  float v[4];
  float s = 0.f;
  for (int j = 0; j < 4; ++j) {
    int c = tid + j * 256;
    v[j] = x[base + c] + res[base + c];
    s += v[j];
  }
  red[tid] = s; __syncthreads();
  for (int o = 128; o > 0; o >>= 1) {
    if (tid < o) red[tid] += red[tid + o];
    __syncthreads();
  }
  float mu = red[0] * (1.f / 1024.f);
  __syncthreads();

  s = 0.f;
  for (int j = 0; j < 4; ++j) { float d = v[j] - mu; s += d * d; }
  red[tid] = s; __syncthreads();
  for (int o = 128; o > 0; o >>= 1) {
    if (tid < o) red[tid] += red[tid + o];
    __syncthreads();
  }
  float rstd = rsqrtf(red[0] * (1.f / 1024.f) + 1e-5f);

  for (int j = 0; j < 4; ++j) {
    int c = tid + j * 256;
    float o_ = (v[j] - mu) * rstd * g[c] + b[c];
    y[base + c] = o_;
    if (ybf) ybf[base + c] = f32_to_bf16(o_);
  }
}

// ---------------------------------------------------------------------------
// Host-side orchestration
// ---------------------------------------------------------------------------
extern "C" void kernel_launch(void* const* d_in, const int* in_sizes, int n_in,
                              void* d_out, int out_size, void* d_ws, size_t ws_size,
                              hipStream_t stream) {
  (void)in_sizes; (void)n_in; (void)out_size; (void)ws_size;
  constexpr int Bq = 2, L = 2048, S = 2048, D = 1024, H = 16, FF = 4096;
  constexpr size_t ML = (size_t)Bq * L;  // 4096 rows
  const float inv_temp = 1.0f / 32.0f;   // 1/sqrt(D)

  const float* tgt        = (const float*)d_in[0];
  const float* src        = (const float*)d_in[1];
  /* d_in[2] tgt_mask: causal, computed analytically */
  const float* attn_bias  = (const float*)d_in[3];   // (1,H,L,L)
  const float* attn2_bias = (const float*)d_in[4];   // (1,H,L,S)
  const float* sa_wq = (const float*)d_in[5];  const float* sa_bq = (const float*)d_in[6];
  const float* sa_wk = (const float*)d_in[7];  const float* sa_bk = (const float*)d_in[8];
  const float* sa_wv = (const float*)d_in[9];  const float* sa_bv = (const float*)d_in[10];
  const float* sa_wo = (const float*)d_in[11]; const float* sa_bo = (const float*)d_in[12];
  const float* sa_g  = (const float*)d_in[13]; const float* sa_b  = (const float*)d_in[14];
  const float* ca_wq = (const float*)d_in[15]; const float* ca_bq = (const float*)d_in[16];
  const float* ca_wk = (const float*)d_in[17]; const float* ca_bk = (const float*)d_in[18];
  const float* ca_wv = (const float*)d_in[19]; const float* ca_bv = (const float*)d_in[20];
  const float* ca_wo = (const float*)d_in[21]; const float* ca_bo = (const float*)d_in[22];
  const float* ca_g  = (const float*)d_in[23]; const float* ca_b  = (const float*)d_in[24];
  const float* ff_w1 = (const float*)d_in[25]; const float* ff_b1 = (const float*)d_in[26];
  const float* ff_w2 = (const float*)d_in[27]; const float* ff_b2 = (const float*)d_in[28];
  const float* ff_g  = (const float*)d_in[29]; const float* ff_b  = (const float*)d_in[30];
  float* out = (float*)d_out;

  char* p = (char*)d_ws;
  auto take = [&](size_t bytes) -> void* {
    void* r = (void*)p;
    p += (bytes + 255) & ~(size_t)255;
    return r;
  };
  typedef unsigned short us;
  us* tgt_bf  = (us*)take(ML * D * 2);
  us* src_bf  = (us*)take(ML * D * 2);
  us* w_sa_q  = (us*)take((size_t)D * D * 2);
  us* w_sa_k  = (us*)take((size_t)D * D * 2);
  us* w_sa_v  = (us*)take((size_t)D * D * 2);
  us* w_sa_o  = (us*)take((size_t)D * D * 2);
  us* w_ca_q  = (us*)take((size_t)D * D * 2);
  us* w_ca_k  = (us*)take((size_t)D * D * 2);
  us* w_ca_v  = (us*)take((size_t)D * D * 2);
  us* w_ca_o  = (us*)take((size_t)D * D * 2);
  us* w_ff_1  = (us*)take((size_t)D * FF * 2);
  us* w_ff_2  = (us*)take((size_t)FF * D * 2);
  us* q_bf    = (us*)take(ML * D * 2);
  us* k_bf    = (us*)take(ML * D * 2);
  us* v_bf    = (us*)take(ML * D * 2);
  us* att_bf  = (us*)take(ML * D * 2);
  float* proj32 = (float*)take(ML * D * 4);
  float* out1   = (float*)take(ML * D * 4);
  float* out2   = (float*)take(ML * D * 4);
  us* out2_bf   = (us*)take(ML * D * 2);
  us* h_bf      = (us*)take(ML * FF * 2);

  auto cvt = [&](const float* in, us* o, size_t n) {
    cvt_bf16_kernel<<<1024, 256, 0, stream>>>(in, o, n);
  };
  auto gemm = [&](const us* A, const us* Bm, const float* bias, float* C32,
                  us* C16, int Mm, int Nn, int Kk, int relu) {
    dim3 g(Nn / 128, Mm / 128);
    gemm_bf16_kernel<<<g, 256, 0, stream>>>(A, Bm, bias, C32, C16, Mm, Nn, Kk, relu);
  };

  cvt(tgt, tgt_bf, ML * D);
  cvt(src, src_bf, ML * D);
  cvt(sa_wq, w_sa_q, (size_t)D * D);  cvt(sa_wk, w_sa_k, (size_t)D * D);
  cvt(sa_wv, w_sa_v, (size_t)D * D);  cvt(sa_wo, w_sa_o, (size_t)D * D);
  cvt(ca_wq, w_ca_q, (size_t)D * D);  cvt(ca_wk, w_ca_k, (size_t)D * D);
  cvt(ca_wv, w_ca_v, (size_t)D * D);  cvt(ca_wo, w_ca_o, (size_t)D * D);
  cvt(ff_w1, w_ff_1, (size_t)D * FF); cvt(ff_w2, w_ff_2, (size_t)FF * D);

  dim3 ga(L / 64, H, Bq);

  // ---- self attention ----
  gemm(tgt_bf, w_sa_q, sa_bq, nullptr, q_bf, (int)ML, D, D, 0);
  gemm(tgt_bf, w_sa_k, sa_bk, nullptr, k_bf, (int)ML, D, D, 0);
  gemm(tgt_bf, w_sa_v, sa_bv, nullptr, v_bf, (int)ML, D, D, 0);
  flash_attn_kernel<true><<<ga, 128, 0, stream>>>(q_bf, k_bf, v_bf, attn_bias,
                                                  att_bf, L, L, inv_temp);
  gemm(att_bf, w_sa_o, sa_bo, proj32, nullptr, (int)ML, D, D, 0);
  add_ln_kernel<<<(int)ML, 256, 0, stream>>>(proj32, tgt, sa_g, sa_b, out1, nullptr);

  // ---- cross attention (query = raw tgt, faithful to reference) ----
  gemm(tgt_bf, w_ca_q, ca_bq, nullptr, q_bf, (int)ML, D, D, 0);
  gemm(src_bf, w_ca_k, ca_bk, nullptr, k_bf, (int)ML, D, D, 0);
  gemm(src_bf, w_ca_v, ca_bv, nullptr, v_bf, (int)ML, D, D, 0);
  flash_attn_kernel<false><<<ga, 128, 0, stream>>>(q_bf, k_bf, v_bf, attn2_bias,
                                                   att_bf, L, S, inv_temp);
  gemm(att_bf, w_ca_o, ca_bo, proj32, nullptr, (int)ML, D, D, 0);
  add_ln_kernel<<<(int)ML, 256, 0, stream>>>(proj32, out1, ca_g, ca_b, out2, out2_bf);

  // ---- feedforward: relu(relu(x@w1+b1)@w2+b2), then LN(ff + x) ----
  gemm(out2_bf, w_ff_1, ff_b1, nullptr, h_bf, (int)ML, FF, D, 1);
  gemm(h_bf, w_ff_2, ff_b2, proj32, nullptr, (int)ML, D, FF, 1);
  add_ln_kernel<<<(int)ML, 256, 0, stream>>>(proj32, out2, ff_g, ff_b, out, nullptr);
}
